// Decoder_89292370084531
// MI455X (gfx1250) — matrix operand order repaired and checked by reading, not compiled, not run on hardware
//
#include <hip/hip_runtime.h>
#include <cstdint>
#include <cstddef>

// ---------------- constants ----------------
#define B_    32
#define S_    1000
#define T_    200      // S/SHRINK
#define SRC_  256
#define DIM_  1024
#define E_    128
#define FF_   2048
#define HALF_ 512
#define V_    256
#define NL_   6
#define NEG_  (-1e9f)

typedef __attribute__((ext_vector_type(16))) __bf16          v16bf;
typedef __attribute__((ext_vector_type(8)))  float           v8f;
typedef __attribute__((ext_vector_type(16))) unsigned short  v16u;

union Frag { unsigned short s[16]; uint4 q[2]; v16u v; };

__device__ __forceinline__ unsigned short f2bf(float f) {
  union { float f; unsigned int u; } x; x.f = f;
  unsigned int lsb = (x.u >> 16) & 1u;
  x.u += 0x7fffu + lsb;
  return (unsigned short)(x.u >> 16);
}

__device__ __forceinline__ float sigf(float x) { return 1.f / (1.f + __expf(-x)); }

__device__ __forceinline__ v8f wmma_bf16(const Frag& a, const Frag& b, v8f c) {
  v16bf av = __builtin_bit_cast(v16bf, a.v);
  v16bf bv = __builtin_bit_cast(v16bf, b.v);
  return __builtin_amdgcn_wmma_f32_16x16x32_bf16(false, av, false, bv, (short)0, c,
                                                 false, false);
}

// Async copy: 16B per lane, mem -> LDS, tracked by ASYNCcnt.
// INST_OFFSET is applied to BOTH the LDS and global address (ISA 08 §4.4),
// so one base pair serves two 16B chunks.
__device__ __forceinline__ void async_cp32(unsigned lds_off, const void* gptr) {
  unsigned long long g = (unsigned long long)gptr;
  asm volatile("global_load_async_to_lds_b128 %0, %1, off"
               :: "v"(lds_off), "v"(g) : "memory");
  asm volatile("global_load_async_to_lds_b128 %0, %1, off offset:16"
               :: "v"(lds_off), "v"(g) : "memory");
}
__device__ __forceinline__ void wait_async0() {
  asm volatile("s_wait_asynccnt 0" ::: "memory");
}

enum { GF_RELU = 1, GF_ACC = 2, GF_BIAS = 4, GF_WRBF = 8 };

// ---------------- generic bf16 WMMA GEMM ----------------
// C[M,N] f32 = A[M,K]bf16 * B[K,N]bf16 (+bias)(+C)(relu), optional bf16 copy.
// Requires: M%128==0, N%128==0, K%32==0.
// Double-buffered LDS; A tile staged with GLOBAL_LOAD_ASYNC_TO_LDS_B128,
// B tile register-transposed into [col][k] so WMMA B fragments are contiguous.
__global__ __launch_bounds__(256) void gemm_bf16_kernel(
    const unsigned short* __restrict__ A, int lda,
    const unsigned short* __restrict__ Bm, int ldb,
    float* __restrict__ C, int ldc,
    const float* __restrict__ bias,
    unsigned short* __restrict__ Cbf, int ldbf,
    int K, int flags)
{
  __shared__ unsigned short As[2][128 * 32];   // [buf][row][k]
  __shared__ unsigned short Bs[2][128 * 32];   // [buf][col][k] (transposed)
  const int row0 = blockIdx.y * 128, col0 = blockIdx.x * 128;
  const int tid  = threadIdx.x;
  const int wave = tid >> 5, lane = tid & 31;
  const int wm = wave >> 1, wn = wave & 1;      // 4 x 2 wave grid
  const int half = lane >> 4, ln16 = lane & 15;

  const v8f vzero = {0.f, 0.f, 0.f, 0.f, 0.f, 0.f, 0.f, 0.f};
  v8f acc[2][4];
#pragma unroll
  for (int i = 0; i < 2; ++i)
#pragma unroll
    for (int j = 0; j < 4; ++j) acc[i][j] = vzero;

  const int ar = tid >> 1, ac = (tid & 1) * 16;      // A tile: 128 rows x 32 k
  const int bkr = tid >> 3, bnc = (tid & 7) * 16;    // B tile: 32 k x 128 cols

  const unsigned asA[2] = { (unsigned)(size_t)&As[0][ar * 32 + ac],
                            (unsigned)(size_t)&As[1][ar * 32 + ac] };

  const int nk = K / 32;
  Frag bt;

  // prologue: stage tile 0
  async_cp32(asA[0], A + (size_t)(row0 + ar) * lda + ac);
  {
    const unsigned short* bp = Bm + (size_t)bkr * ldb + col0 + bnc;
    bt.q[0] = *(const uint4*)bp;
    bt.q[1] = *(const uint4*)(bp + 8);
#pragma unroll
    for (int j = 0; j < 16; ++j) Bs[0][(bnc + j) * 32 + bkr] = bt.s[j];
  }

  for (int i = 0; i < nk; ++i) {
    const int cur = i & 1;
    wait_async0();          // this wave's async tile copies have landed in LDS
    __syncthreads();        // (compiler adds s_wait_dscnt before the barrier)

    // kick off next tile: async A copy + B register fetch (overlaps WMMAs)
    if (i + 1 < nk) {
      const int kk = (i + 1) * 32;
      async_cp32(asA[cur ^ 1], A + (size_t)(row0 + ar) * lda + kk + ac);
      const unsigned short* bp = Bm + (size_t)(kk + bkr) * ldb + col0 + bnc;
      if (i + 2 < nk) __builtin_prefetch(bp + (size_t)32 * ldb, 0, 0);
      bt.q[0] = *(const uint4*)bp;
      bt.q[1] = *(const uint4*)(bp + 8);
    }

    // fragments from current buffers
    const unsigned short* asb = As[cur];
    const unsigned short* bsb = Bs[cur];
    Frag af[2], bfr[4];
#pragma unroll
    for (int mi = 0; mi < 2; ++mi) {
      // ISA 16-bit A layout: lanes0-15 K={0..7,16..23}, lanes16-31 +8
      const unsigned short* ap = &asb[(wm * 32 + mi * 16 + ln16) * 32 + half * 8];
      af[mi].q[0] = *(const uint4*)ap;
      af[mi].q[1] = *(const uint4*)(ap + 16);
    }
#pragma unroll
    for (int ni = 0; ni < 4; ++ni) {
      // B: lane = column, lanes0-15 K={0..7,16..23}, lanes16-31 +8
      const unsigned short* bp = &bsb[(wn * 64 + ni * 16 + ln16) * 32 + half * 8];
      bfr[ni].q[0] = *(const uint4*)bp;
      bfr[ni].q[1] = *(const uint4*)(bp + 16);
    }
#pragma unroll
    for (int mi = 0; mi < 2; ++mi)
#pragma unroll
      for (int ni = 0; ni < 4; ++ni)
        acc[mi][ni] = wmma_bf16(af[mi], bfr[ni], acc[mi][ni]);

    // scatter next B tile into the alternate buffer (safe: nobody reads it
    // until after the next barrier)
    if (i + 1 < nk) {
#pragma unroll
      for (int j = 0; j < 16; ++j) Bs[cur ^ 1][(bnc + j) * 32 + bkr] = bt.s[j];
    }
  }

#pragma unroll
  for (int mi = 0; mi < 2; ++mi) {
#pragma unroll
    for (int ni = 0; ni < 4; ++ni) {
      const int gcol  = col0 + wn * 64 + ni * 16 + ln16;
      const int rbase = row0 + wm * 32 + mi * 16 + half * 8;
      const float bv  = (flags & GF_BIAS) ? bias[gcol] : 0.f;
#pragma unroll
      for (int r = 0; r < 8; ++r) {
        const size_t ci = (size_t)(rbase + r) * ldc + gcol;
        float v = acc[mi][ni][r] + bv;
        if (flags & GF_ACC)  v += C[ci];
        if (flags & GF_RELU) v = fmaxf(v, 0.f);
        C[ci] = v;
        if (flags & GF_WRBF) Cbf[(size_t)(rbase + r) * ldbf + gcol] = f2bf(v);
      }
    }
  }
}

// ---------------- LSTM recurrence GEMM (M=32) ----------------
// gates[32,4096] = Xt[32,640] @ WihT[640,4096] + H[32,1024] @ WhhT[1024,4096]
__global__ __launch_bounds__(256) void lstm_gemm_kernel(
    const unsigned short* __restrict__ Xt, int ldx,
    const unsigned short* __restrict__ WihT,
    const unsigned short* __restrict__ Hbf,
    const unsigned short* __restrict__ WhhT,
    float* __restrict__ gates)
{
  const int col0 = blockIdx.x * 128;
  const int wave = threadIdx.x >> 5, lane = threadIdx.x & 31;
  const int half = lane >> 4, ln16 = lane & 15;
  const int col  = col0 + wave * 16 + ln16;

  const v8f vzero = {0.f, 0.f, 0.f, 0.f, 0.f, 0.f, 0.f, 0.f};
  v8f acc0 = vzero, acc1 = vzero;

  for (int kk = 0; kk < 640; kk += 32) {
    Frag a0, a1, bfr;
    const unsigned short* ap0 = Xt + (size_t)ln16 * ldx + kk + half * 8;
    a0.q[0] = *(const uint4*)ap0;  a0.q[1] = *(const uint4*)(ap0 + 16);
    const unsigned short* ap1 = Xt + (size_t)(16 + ln16) * ldx + kk + half * 8;
    a1.q[0] = *(const uint4*)ap1;  a1.q[1] = *(const uint4*)(ap1 + 16);
#pragma unroll
    for (int e = 0; e < 16; ++e)
      bfr.s[e] = WihT[(size_t)(kk + (e >> 3) * 16 + half * 8 + (e & 7)) * (4 * DIM_) + col];
    acc0 = wmma_bf16(a0, bfr, acc0);
    acc1 = wmma_bf16(a1, bfr, acc1);
  }
  for (int kk = 0; kk < 1024; kk += 32) {
    Frag a0, a1, bfr;
    const unsigned short* ap0 = Hbf + (size_t)ln16 * DIM_ + kk + half * 8;
    a0.q[0] = *(const uint4*)ap0;  a0.q[1] = *(const uint4*)(ap0 + 16);
    const unsigned short* ap1 = Hbf + (size_t)(16 + ln16) * DIM_ + kk + half * 8;
    a1.q[0] = *(const uint4*)ap1;  a1.q[1] = *(const uint4*)(ap1 + 16);
#pragma unroll
    for (int e = 0; e < 16; ++e)
      bfr.s[e] = WhhT[(size_t)(kk + (e >> 3) * 16 + half * 8 + (e & 7)) * (4 * DIM_) + col];
    acc0 = wmma_bf16(a0, bfr, acc0);
    acc1 = wmma_bf16(a1, bfr, acc1);
  }
#pragma unroll
  for (int r = 0; r < 8; ++r) {
    gates[(size_t)(half * 8 + r) * (4 * DIM_) + col]      = acc0[r];
    gates[(size_t)(16 + half * 8 + r) * (4 * DIM_) + col] = acc1[r];
  }
}

// ---------------- LSTM gate / state update ----------------
__global__ __launch_bounds__(256) void lstm_gate_kernel(
    const float* __restrict__ gates, const float* __restrict__ b_ih,
    const float* __restrict__ b_hh, float* __restrict__ c,
    unsigned short* __restrict__ hbf, unsigned short* __restrict__ lstm_bf, int t)
{
  const int idx = blockIdx.x * 256 + threadIdx.x;
  if (idx >= B_ * DIM_) return;
  const int b = idx >> 10, j = idx & 1023;
  const float* gp = gates + (size_t)b * 4 * DIM_;
  const float gi = gp[j]            + b_ih[j]            + b_hh[j];
  const float gf = gp[DIM_ + j]     + b_ih[DIM_ + j]     + b_hh[DIM_ + j];
  const float gg = gp[2 * DIM_ + j] + b_ih[2 * DIM_ + j] + b_hh[2 * DIM_ + j];
  const float go = gp[3 * DIM_ + j] + b_ih[3 * DIM_ + j] + b_hh[3 * DIM_ + j];
  const float cc = sigf(gf) * c[idx] + sigf(gi) * tanhf(gg);
  const float hh = sigf(go) * tanhf(cc);
  c[idx] = cc;
  hbf[idx] = f2bf(hh);
  lstm_bf[((size_t)b * S_ + t) * DIM_ + j] = f2bf(hh);
}

// ---------------- attention (VALU path, <6% of FLOPs) ----------------
__global__ __launch_bounds__(64) void self_attn_kernel(
    const float* __restrict__ qkv, const float* __restrict__ smask,
    unsigned short* __restrict__ attn_bf)
{
  const int q = blockIdx.x, h = blockIdx.y, b = blockIdx.z;
  const int tid = threadIdx.x;
  __shared__ float sc[T_];
  __shared__ float red[64];
  const float* qv = qkv + ((size_t)(b * T_ + q)) * (3 * DIM_) + h * 64;
  float lmax = -3e38f;
  for (int k = tid; k < T_; k += 64) {
    const float* kp = qkv + ((size_t)(b * T_ + k)) * (3 * DIM_) + DIM_ + h * 64;
    float d = 0.f;
#pragma unroll
    for (int e = 0; e < 64; ++e) d += qv[e] * kp[e];
    float s = d * 0.125f;
    if (k > q) s += NEG_;
    s += (1.f - smask[b * T_ + k]) * NEG_;
    sc[k] = s;
    lmax = fmaxf(lmax, s);
  }
  red[tid] = lmax; __syncthreads();
  for (int st = 32; st > 0; st >>= 1) { if (tid < st) red[tid] = fmaxf(red[tid], red[tid + st]); __syncthreads(); }
  const float m = red[0]; __syncthreads();
  float lsum = 0.f;
  for (int k = tid; k < T_; k += 64) { float e = __expf(sc[k] - m); sc[k] = e; lsum += e; }
  red[tid] = lsum; __syncthreads();
  for (int st = 32; st > 0; st >>= 1) { if (tid < st) red[tid] += red[tid + st]; __syncthreads(); }
  const float inv = 1.f / red[0];
  float o = 0.f;
  for (int k = 0; k < T_; ++k)
    o += sc[k] * qkv[((size_t)(b * T_ + k)) * (3 * DIM_) + 2 * DIM_ + h * 64 + tid];
  attn_bf[((size_t)(b * T_ + q)) * DIM_ + h * 64 + tid] = f2bf(o * inv);
}

__global__ __launch_bounds__(64) void cross_attn_kernel(
    const float* __restrict__ qc, const float* __restrict__ kvc,
    const float* __restrict__ enc_mask, unsigned short* __restrict__ attn_bf)
{
  const int q = blockIdx.x, h = blockIdx.y, b = blockIdx.z;
  const int tid = threadIdx.x;
  __shared__ float sc[SRC_];
  __shared__ float red[64];
  const float* qv = qc + ((size_t)(b * T_ + q)) * DIM_ + h * 64;
  float lmax = -3e38f;
  for (int k = tid; k < SRC_; k += 64) {
    const float* kp = kvc + ((size_t)(b * SRC_ + k)) * (2 * DIM_) + h * 64;
    float d = 0.f;
#pragma unroll
    for (int e = 0; e < 64; ++e) d += qv[e] * kp[e];
    float s = d * 0.125f + (1.f - enc_mask[b * SRC_ + k]) * NEG_;
    sc[k] = s;
    lmax = fmaxf(lmax, s);
  }
  red[tid] = lmax; __syncthreads();
  for (int st = 32; st > 0; st >>= 1) { if (tid < st) red[tid] = fmaxf(red[tid], red[tid + st]); __syncthreads(); }
  const float m = red[0]; __syncthreads();
  float lsum = 0.f;
  for (int k = tid; k < SRC_; k += 64) { float e = __expf(sc[k] - m); sc[k] = e; lsum += e; }
  red[tid] = lsum; __syncthreads();
  for (int st = 32; st > 0; st >>= 1) { if (tid < st) red[tid] += red[tid + st]; __syncthreads(); }
  const float inv = 1.f / red[0];
  float o = 0.f;
  for (int k = 0; k < SRC_; ++k)
    o += sc[k] * kvc[((size_t)(b * SRC_ + k)) * (2 * DIM_) + DIM_ + h * 64 + tid];
  attn_bf[((size_t)(b * T_ + q)) * DIM_ + h * 64 + tid] = f2bf(o * inv);
}

// ---------------- LayerNorm (optional residual, writes f32 + bf16) ----------------
__global__ __launch_bounds__(256) void ln_kernel(
    const float* __restrict__ xin, const float* __restrict__ res,
    const float* __restrict__ gw, const float* __restrict__ bw,
    float* __restrict__ y, unsigned short* __restrict__ ybf, int D)
{
  const int row = blockIdx.x, tid = threadIdx.x;
  __shared__ float red[256];
  const float* xp = xin + (size_t)row * D;
  const float* rp = res ? res + (size_t)row * D : nullptr;
  float s = 0.f;
  for (int d = tid; d < D; d += 256) { float v = xp[d] + (rp ? rp[d] : 0.f); s += v; }
  red[tid] = s; __syncthreads();
  for (int st = 128; st > 0; st >>= 1) { if (tid < st) red[tid] += red[tid + st]; __syncthreads(); }
  const float mean = red[0] / D; __syncthreads();
  float s2 = 0.f;
  for (int d = tid; d < D; d += 256) { float v = xp[d] + (rp ? rp[d] : 0.f) - mean; s2 += v * v; }
  red[tid] = s2; __syncthreads();
  for (int st = 128; st > 0; st >>= 1) { if (tid < st) red[tid] += red[tid + st]; __syncthreads(); }
  const float inv = rsqrtf(red[0] / D + 1e-12f);
  for (int d = tid; d < D; d += 256) {
    float v = (xp[d] + (rp ? rp[d] : 0.f) - mean) * inv * gw[d] + bw[d];
    y[(size_t)row * D + d] = v;
    if (ybf) ybf[(size_t)row * D + d] = f2bf(v);
  }
}

// ---------------- front-end kernels ----------------
__global__ void embed_kernel(const int* __restrict__ tids,
                             const float* __restrict__ cemb,
                             const float* __restrict__ pos,
                             unsigned short* __restrict__ emb_bf)
{
  const int idx = blockIdx.x * 256 + threadIdx.x;
  if (idx >= B_ * S_ * E_) return;
  const int e = idx & 127, row = idx >> 7;
  const int b = row / S_, s = row % S_;
  const int dec = (s < 5) ? 1 : tids[b * S_ + s - 5];
  emb_bf[idx] = f2bf(cemb[dec * E_ + e] + pos[s * E_ + e]);
}

__global__ __launch_bounds__(256) void pool_kernel(
    const float* __restrict__ h_char, const float* __restrict__ tmask,
    const float* __restrict__ bert_pos, const float* __restrict__ bert_tok,
    float* __restrict__ prex, float* __restrict__ smask)
{
  const int row = blockIdx.x;           // b*T + t
  const int b = row / T_, t = row % T_;
  const int tid = threadIdx.x;
  for (int d = tid; d < DIM_; d += 256) {
    float mx = -3e38f;
#pragma unroll
    for (int j = 0; j < 5; ++j) {
      const int s = t * 5 + j;
      const float msk = (s < 5) ? 1.f : tmask[b * S_ + s - 5];
      mx = fmaxf(mx, h_char[((size_t)(b * S_ + s)) * DIM_ + d] * msk);
    }
    prex[(size_t)row * DIM_ + d] = mx + bert_pos[t * DIM_ + d] + bert_tok[d];
  }
  if (tid == 0) {
    float mm = 0.f;
#pragma unroll
    for (int j = 0; j < 5; ++j) {
      const int s = t * 5 + j;
      mm = fmaxf(mm, (s < 5) ? 1.f : tmask[b * S_ + s - 5]);
    }
    smask[row] = mm;
  }
}

__global__ void build_lstmA_kernel(const int* __restrict__ tids,
                                   const float* __restrict__ cemb,
                                   const float* __restrict__ cs,
                                   unsigned short* __restrict__ lstmA)
{
  const long long idx = (long long)blockIdx.x * 256 + threadIdx.x;
  if (idx >= (long long)B_ * S_ * 640) return;
  const int row = (int)(idx / 640), col = (int)(idx % 640);
  const int b = row / S_, s = row % S_;
  float v;
  if (col < E_) {
    const int pad = (s == 0) ? 1 : tids[b * S_ + s - 1];
    v = cemb[pad * E_ + col];
  } else {
    v = cs[(size_t)row * HALF_ + (col - E_)];
  }
  lstmA[idx] = f2bf(v);
}

// ---------------- NLL ----------------
__global__ void nll_kernel(const float* __restrict__ logits,
                           const int* __restrict__ tids,
                           const float* __restrict__ tmask,
                           float* __restrict__ sums)
{
  const int row = blockIdx.x * 256 + threadIdx.x;
  if (row >= B_ * S_) return;
  const float* lp = logits + (size_t)row * V_;
  float m = lp[0];
  for (int j = 1; j < V_; ++j) m = fmaxf(m, lp[j]);
  float s = 0.f;
  for (int j = 0; j < V_; ++j) s += __expf(lp[j] - m);
  const float nll = (m + __logf(s)) - lp[tids[row]];
  const float w = tmask[row];
  atomicAdd(&sums[0], nll * w);
  atomicAdd(&sums[1], w);
}

__global__ void finalize_kernel(const float* __restrict__ sums, float* __restrict__ out)
{
  if (threadIdx.x == 0 && blockIdx.x == 0) out[0] = sums[0] / sums[1];
}

__global__ void init_kernel(float* __restrict__ c, unsigned short* __restrict__ hbf,
                            float* __restrict__ sums)
{
  const int idx = blockIdx.x * 256 + threadIdx.x;
  if (idx < B_ * DIM_) { c[idx] = 0.f; hbf[idx] = 0; }
  if (idx < 2) sums[idx] = 0.f;
}

// ---------------- conversions ----------------
__global__ void cvt_bf_kernel(const float* __restrict__ s, unsigned short* __restrict__ d,
                              long long n)
{
  const long long i = (long long)blockIdx.x * 256 + threadIdx.x;
  if (i < n) d[i] = f2bf(s[i]);
}

__global__ void cvtT_bf_kernel(const float* __restrict__ s, unsigned short* __restrict__ d,
                               int R, int C)
{
  const long long i = (long long)blockIdx.x * 256 + threadIdx.x;
  if (i < (long long)R * C) {
    const int r = (int)(i / C), c = (int)(i % C);
    d[(size_t)c * R + r] = f2bf(s[i]);
  }
}

// ---------------- host orchestration ----------------
extern "C" void kernel_launch(void* const* d_in, const int* in_sizes, int n_in,
                              void* d_out, int out_size, void* d_ws, size_t ws_size,
                              hipStream_t stream)
{
  const float* encoder_states = (const float*)d_in[0];
  const float* encoder_mask   = (const float*)d_in[1];
  const int*   target_ids     = (const int*)d_in[2];
  const float* target_mask    = (const float*)d_in[3];
  const float* char_emb_w     = (const float*)d_in[4];
  const float* pre_pos_emb    = (const float*)d_in[5];
  const float* Wc   = (const float*)d_in[6];
  const float* bc   = (const float*)d_in[7];
  const float* bert_pos = (const float*)d_in[8];
  const float* bert_tok = (const float*)d_in[9];
  const float* emb_ln_g = (const float*)d_in[10];
  const float* emb_ln_b = (const float*)d_in[11];
  const float* Wqkv = (const float*)d_in[12];
  const float* bqkv = (const float*)d_in[13];
  const float* Wo   = (const float*)d_in[14];
  const float* bo   = (const float*)d_in[15];
  const float* ln1g = (const float*)d_in[16];
  const float* ln1b = (const float*)d_in[17];
  const float* Cq   = (const float*)d_in[18];
  const float* cbq  = (const float*)d_in[19];
  const float* Ckv  = (const float*)d_in[20];
  const float* cbkv = (const float*)d_in[21];
  const float* Co   = (const float*)d_in[22];
  const float* cbo  = (const float*)d_in[23];
  const float* ln2g = (const float*)d_in[24];
  const float* ln2b = (const float*)d_in[25];
  const float* W1   = (const float*)d_in[26];
  const float* b1   = (const float*)d_in[27];
  const float* W2   = (const float*)d_in[28];
  const float* b2   = (const float*)d_in[29];
  const float* ln3g = (const float*)d_in[30];
  const float* ln3b = (const float*)d_in[31];
  const float* Wnar = (const float*)d_in[32];
  const float* bnar = (const float*)d_in[33];
  const float* W_ih = (const float*)d_in[34];
  const float* W_hh = (const float*)d_in[35];
  const float* b_ih = (const float*)d_in[36];
  const float* b_hh = (const float*)d_in[37];
  const float* Wout = (const float*)d_in[38];
  const float* bout = (const float*)d_in[39];

  // workspace bump allocator
  char* p = (char*)d_ws;
  auto alloc_us = [&](size_t n) -> unsigned short* {
    unsigned short* r = (unsigned short*)p;
    p += ((n * 2 + 255) & ~(size_t)255);
    return r;
  };
  auto alloc_f = [&](size_t n) -> float* {
    float* r = (float*)p;
    p += ((n * 4 + 255) & ~(size_t)255);
    return r;
  };

  // bf16 weights & activations
  unsigned short* enc_bf  = alloc_us((size_t)B_ * SRC_ * DIM_);
  unsigned short* Wc_bf   = alloc_us((size_t)E_ * DIM_);
  unsigned short* Wqkv_bf = alloc_us((size_t)NL_ * DIM_ * 3 * DIM_);
  unsigned short* Wo_bf   = alloc_us((size_t)NL_ * DIM_ * DIM_);
  unsigned short* Cq_bf   = alloc_us((size_t)NL_ * DIM_ * DIM_);
  unsigned short* Ckv_bf  = alloc_us((size_t)NL_ * DIM_ * 2 * DIM_);
  unsigned short* Co_bf   = alloc_us((size_t)NL_ * DIM_ * DIM_);
  unsigned short* W1_bf   = alloc_us((size_t)NL_ * DIM_ * FF_);
  unsigned short* W2_bf   = alloc_us((size_t)NL_ * FF_ * DIM_);
  unsigned short* Wnar_bf = alloc_us((size_t)DIM_ * 5 * HALF_);
  unsigned short* WihT_bf = alloc_us((size_t)640 * 4 * DIM_);
  unsigned short* WhhT_bf = alloc_us((size_t)DIM_ * 4 * DIM_);
  unsigned short* Wout_bf = alloc_us((size_t)(DIM_ + HALF_) * V_);
  unsigned short* emb_bf  = alloc_us((size_t)B_ * S_ * E_);
  unsigned short* x_bf    = alloc_us((size_t)B_ * T_ * DIM_);
  unsigned short* attn_bf = alloc_us((size_t)B_ * T_ * DIM_);
  unsigned short* ff_bf   = alloc_us((size_t)B_ * T_ * FF_);
  unsigned short* lstmA   = alloc_us((size_t)B_ * S_ * 640);
  unsigned short* h_bf    = alloc_us((size_t)B_ * DIM_);
  unsigned short* lstm_bf = alloc_us((size_t)B_ * S_ * DIM_);

  // f32 activations
  float* h_char = alloc_f((size_t)B_ * S_ * DIM_);
  float* prex   = alloc_f((size_t)B_ * T_ * DIM_);
  float* smask  = alloc_f((size_t)B_ * T_);
  float* xbuf   = alloc_f((size_t)B_ * T_ * DIM_);
  float* qkv    = alloc_f((size_t)B_ * T_ * 3 * DIM_);
  float* qc     = alloc_f((size_t)B_ * T_ * DIM_);
  float* kvc    = alloc_f((size_t)B_ * SRC_ * 2 * DIM_);
  float* tmp    = alloc_f((size_t)B_ * T_ * DIM_);
  float* ffbuf  = alloc_f((size_t)B_ * T_ * FF_);
  float* cs     = alloc_f((size_t)B_ * S_ * HALF_);
  float* gates  = alloc_f((size_t)B_ * 4 * DIM_);
  float* cvec   = alloc_f((size_t)B_ * DIM_);
  float* sums   = alloc_f(2);
  float* logits = alloc_f((size_t)B_ * S_ * V_);

  auto cvt = [&](const float* s, unsigned short* d, size_t n) {
    cvt_bf_kernel<<<(unsigned)((n + 255) / 256), 256, 0, stream>>>(s, d, (long long)n);
  };
  auto gemm = [&](const unsigned short* A, int lda, const unsigned short* Bm, int ldb,
                  float* C, int ldc, const float* bias, unsigned short* Cbf, int ldbf,
                  int M, int N, int K, int flags) {
    dim3 g(N / 128, M / 128);
    gemm_bf16_kernel<<<g, 256, 0, stream>>>(A, lda, Bm, ldb, C, ldc, bias, Cbf, ldbf, K, flags);
  };

  // 0) init LSTM state + accumulators
  init_kernel<<<(B_ * DIM_ + 255) / 256, 256, 0, stream>>>(cvec, h_bf, sums);

  // 1) weight / encoder conversions
  cvt(encoder_states, enc_bf, (size_t)B_ * SRC_ * DIM_);
  cvt(Wc, Wc_bf, (size_t)E_ * DIM_);
  cvt(Wqkv, Wqkv_bf, (size_t)NL_ * DIM_ * 3 * DIM_);
  cvt(Wo, Wo_bf, (size_t)NL_ * DIM_ * DIM_);
  cvt(Cq, Cq_bf, (size_t)NL_ * DIM_ * DIM_);
  cvt(Ckv, Ckv_bf, (size_t)NL_ * DIM_ * 2 * DIM_);
  cvt(Co, Co_bf, (size_t)NL_ * DIM_ * DIM_);
  cvt(W1, W1_bf, (size_t)NL_ * DIM_ * FF_);
  cvt(W2, W2_bf, (size_t)NL_ * FF_ * DIM_);
  cvt(Wnar, Wnar_bf, (size_t)DIM_ * 5 * HALF_);
  cvt(Wout, Wout_bf, (size_t)(DIM_ + HALF_) * V_);
  {
    long long n = (long long)(4 * DIM_) * 640;
    cvtT_bf_kernel<<<(unsigned)((n + 255) / 256), 256, 0, stream>>>(W_ih, WihT_bf, 4 * DIM_, 640);
    n = (long long)(4 * DIM_) * DIM_;
    cvtT_bf_kernel<<<(unsigned)((n + 255) / 256), 256, 0, stream>>>(W_hh, WhhT_bf, 4 * DIM_, DIM_);
  }

  // 2) char embedding -> projection -> pool -> bert LN
  embed_kernel<<<(B_ * S_ * E_ + 255) / 256, 256, 0, stream>>>(target_ids, char_emb_w,
                                                               pre_pos_emb, emb_bf);
  gemm(emb_bf, E_, Wc_bf, DIM_, h_char, DIM_, bc, nullptr, 0,
       B_ * S_, DIM_, E_, GF_BIAS | GF_RELU);
  pool_kernel<<<B_ * T_, 256, 0, stream>>>(h_char, target_mask, bert_pos, bert_tok, prex, smask);
  ln_kernel<<<B_ * T_, 256, 0, stream>>>(prex, nullptr, emb_ln_g, emb_ln_b, xbuf, x_bf, DIM_);

  // 3) transformer layers
  for (int l = 0; l < NL_; ++l) {
    // self-attention
    gemm(x_bf, DIM_, Wqkv_bf + (size_t)l * DIM_ * 3 * DIM_, 3 * DIM_,
         qkv, 3 * DIM_, bqkv + (size_t)l * 3 * DIM_, nullptr, 0,
         B_ * T_, 3 * DIM_, DIM_, GF_BIAS);
    self_attn_kernel<<<dim3(T_, 16, B_), 64, 0, stream>>>(qkv, smask, attn_bf);
    gemm(attn_bf, DIM_, Wo_bf + (size_t)l * DIM_ * DIM_, DIM_,
         tmp, DIM_, bo + (size_t)l * DIM_, nullptr, 0,
         B_ * T_, DIM_, DIM_, GF_BIAS);
    ln_kernel<<<B_ * T_, 256, 0, stream>>>(tmp, xbuf, ln1g + (size_t)l * DIM_,
                                           ln1b + (size_t)l * DIM_, xbuf, x_bf, DIM_);
    // cross-attention
    gemm(x_bf, DIM_, Cq_bf + (size_t)l * DIM_ * DIM_, DIM_,
         qc, DIM_, cbq + (size_t)l * DIM_, nullptr, 0,
         B_ * T_, DIM_, DIM_, GF_BIAS);
    gemm(enc_bf, DIM_, Ckv_bf + (size_t)l * DIM_ * 2 * DIM_, 2 * DIM_,
         kvc, 2 * DIM_, cbkv + (size_t)l * 2 * DIM_, nullptr, 0,
         B_ * SRC_, 2 * DIM_, DIM_, GF_BIAS);
    cross_attn_kernel<<<dim3(T_, 16, B_), 64, 0, stream>>>(qc, kvc, encoder_mask, attn_bf);
    gemm(attn_bf, DIM_, Co_bf + (size_t)l * DIM_ * DIM_, DIM_,
         tmp, DIM_, cbo + (size_t)l * DIM_, nullptr, 0,
         B_ * T_, DIM_, DIM_, GF_BIAS);
    ln_kernel<<<B_ * T_, 256, 0, stream>>>(tmp, xbuf, ln2g + (size_t)l * DIM_,
                                           ln2b + (size_t)l * DIM_, xbuf, x_bf, DIM_);
    // feed-forward
    gemm(x_bf, DIM_, W1_bf + (size_t)l * DIM_ * FF_, FF_,
         ffbuf, FF_, b1 + (size_t)l * FF_, ff_bf, FF_,
         B_ * T_, FF_, DIM_, GF_BIAS | GF_RELU | GF_WRBF);
    gemm(ff_bf, FF_, W2_bf + (size_t)l * FF_ * DIM_, DIM_,
         tmp, DIM_, b2 + (size_t)l * DIM_, nullptr, 0,
         B_ * T_, DIM_, FF_, GF_BIAS);
    ln_kernel<<<B_ * T_, 256, 0, stream>>>(tmp, xbuf, ln3g + (size_t)l * DIM_,
                                           ln3b + (size_t)l * DIM_, xbuf, x_bf, DIM_);
  }

  // 4) NAR projection -> per-char states [B,S,HALF] (flat reinterpret of [B*T, 5*HALF])
  gemm(x_bf, DIM_, Wnar_bf, 5 * HALF_, cs, 5 * HALF_, bnar, nullptr, 0,
       B_ * T_, 5 * HALF_, DIM_, GF_BIAS);

  // 5) LSTM input assembly: [char_emb[pad_in] | char_states] as bf16 [B*S, 640]
  {
    long long n = (long long)B_ * S_ * 640;
    build_lstmA_kernel<<<(unsigned)((n + 255) / 256), 256, 0, stream>>>(target_ids, char_emb_w,
                                                                        cs, lstmA);
  }

  // 6) sequential LSTM (WMMA recurrence GEMM + gate kernel per step)
  for (int t = 0; t < S_; ++t) {
    lstm_gemm_kernel<<<(4 * DIM_) / 128, 256, 0, stream>>>(
        lstmA + (size_t)t * 640, S_ * 640, WihT_bf, h_bf, WhhT_bf, gates);
    lstm_gate_kernel<<<(B_ * DIM_ + 255) / 256, 256, 0, stream>>>(
        gates, b_ih, b_hh, cvec, h_bf, lstm_bf, t);
  }

  // 7) logits = [lstm_out | char_states] @ Wout + bout  (two GEMMs, second accumulates)
  gemm(lstm_bf, DIM_, Wout_bf, V_, logits, V_, bout, nullptr, 0,
       B_ * S_, V_, DIM_, GF_BIAS);
  gemm(lstmA + E_, 640, Wout_bf + (size_t)DIM_ * V_, V_, logits, V_, nullptr, nullptr, 0,
       B_ * S_, V_, HALF_, GF_ACC);

  // 8) masked mean NLL
  nll_kernel<<<(B_ * S_ + 255) / 256, 256, 0, stream>>>(logits, target_ids, target_mask, sums);
  finalize_kernel<<<1, 64, 0, stream>>>(sums, (float*)d_out);
}
